// NF4MLPTriton_14121852469689
// MI455X (gfx1250) — matrix operand order repaired and checked by reading, not compiled, not run on hardware
//
#include <hip/hip_runtime.h>

// ---- CDNA5 vector types for WMMA fragments (wave32) ----
typedef __attribute__((ext_vector_type(16))) _Float16 v16h;
typedef __attribute__((ext_vector_type(8)))  _Float16 v8h;
typedef __attribute__((ext_vector_type(8)))  float    v8f;
typedef __attribute__((ext_vector_type(4)))  float    v4f;
typedef __attribute__((ext_vector_type(4)))  int      v4i;

#define D_MODEL 4096
#define HIDDEN  11008
#define MTOT    4096   // BATCH * SEQ
#define QBLK    64     // NF4 scale block

__constant__ float NF4_TAB[16] = {
    -1.0f, -0.6961928009986877f, -0.5250730514526367f, -0.39491748809814453f,
    -0.28444138169288635f, -0.18477343022823334f, -0.09105003625154495f, 0.0f,
    0.07958029955625534f, 0.16093020141124725f, 0.24611230194568634f,
    0.33791524171829224f, 0.44070982933044434f, 0.5626170039176941f,
    0.7229568362236023f, 1.0f};

// Dequantize 16 consecutive int32 NF4 codes into a WMMA f16 B-fragment slice.
// Loads are 4x b128; LUT lives in LDS (bank-broadcast friendly, 16 entries).
__device__ __forceinline__ v16h dequant16(const int* __restrict__ q,
                                          const float* tab, float scale) {
  v4i q0 = *(const v4i*)(q);
  v4i q1 = *(const v4i*)(q + 4);
  v4i q2 = *(const v4i*)(q + 8);
  v4i q3 = *(const v4i*)(q + 12);
  v16h b;
#pragma unroll
  for (int e = 0; e < 4; ++e) {
    b[e]      = (_Float16)(tab[q0[e] & 15] * scale);
    b[4 + e]  = (_Float16)(tab[q1[e] & 15] * scale);
    b[8 + e]  = (_Float16)(tab[q2[e] & 15] * scale);
    b[12 + e] = (_Float16)(tab[q3[e] & 15] * scale);
  }
  return b;
}

// ---------------------------------------------------------------------------
// Kernel 1: h[M,H] = silu(x @ w1^T) * (x @ w2^T), h stored f16.
// Block = 128 threads = 4 waves. Block tile: 64 M x 64 N. Wave tile: 64 M x 16 N
// (4 M-subtiles, dual accumulators sharing one A fragment per WMMA pair).
// ---------------------------------------------------------------------------
__global__ __launch_bounds__(128) void nf4_gateup_kernel(
    const float* __restrict__ x,       // [M, D]
    const int* __restrict__ w1q,       // [H, D] codes
    const float* __restrict__ w1s,     // [H, D/64]
    const int* __restrict__ w2q,       // [H, D]
    const float* __restrict__ w2s,     // [H, D/64]
    _Float16* __restrict__ hout)       // [M, H] f16
{
  __shared__ _Float16 xs[64 * 32];     // x tile (64 M x 32 K) as f16
  __shared__ float    tab[16];

  const int tid  = threadIdx.x;
  const int lane = tid & 31;
  const int wave = tid >> 5;
  if (tid < 16) tab[tid] = NF4_TAB[tid];

  const int m0 = blockIdx.y * 64;
  const int n0 = blockIdx.x * 64 + wave * 16;

  // CDNA5 16-bit WMMA fragment geometry (wave32):
  //  A (16x32): lane&15 = row M; lanes<16 hold K 0..15, lanes>=16 K 16..31 (this code's order)
  //  B (32x16): lane&15 = col N; lanes<16 hold K 0..15, lanes>=16 K 16..31 (this code's order, same as A; not the ISA table order)
  const int frag  = lane & 15;
  const int kgrp  = lane >> 4;
  const int koffA = kgrp * 8;
  const int koffB = kgrp * 16;

  const int*   q1row = w1q + (size_t)(n0 + frag) * D_MODEL;
  const int*   q2row = w2q + (size_t)(n0 + frag) * D_MODEL;
  const float* s1row = w1s + (size_t)(n0 + frag) * (D_MODEL / QBLK);
  const float* s2row = w2s + (size_t)(n0 + frag) * (D_MODEL / QBLK);

  v8f acc1[4], acc2[4];
  v8f vz = {};
#pragma unroll
  for (int s = 0; s < 4; ++s) { acc1[s] = vz; acc2[s] = vz; }

  // Cooperative x staging: each of 128 threads converts 16 consecutive f32 -> f16
  const int sr = tid >> 1;             // tile row 0..63
  const int sc = (tid & 1) * 16;       // tile col 0 or 16
  const float*  xsrc = x + (size_t)(m0 + sr) * D_MODEL + sc;
  _Float16*     xdst = &xs[sr * 32 + sc];

  for (int kb = 0; kb < D_MODEL; kb += 32) {
    __syncthreads();                   // xs reuse barrier (also covers tab init)
    {
      v4f f0 = *(const v4f*)(xsrc + kb);
      v4f f1 = *(const v4f*)(xsrc + kb + 4);
      v4f f2 = *(const v4f*)(xsrc + kb + 8);
      v4f f3 = *(const v4f*)(xsrc + kb + 12);
      v8h h0, h1;
#pragma unroll
      for (int j = 0; j < 4; ++j) {
        h0[j] = (_Float16)f0[j]; h0[4 + j] = (_Float16)f1[j];
        h1[j] = (_Float16)f2[j]; h1[4 + j] = (_Float16)f3[j];
      }
      *(v8h*)(xdst)     = h0;          // ds_store_b128
      *(v8h*)(xdst + 8) = h1;
    }

    // Dequantize B fragments while the LDS stores settle.
    const int sidx = (kb + koffB) >> 6;          // 16 K-values stay in one 64-block
    float sc1 = s1row[sidx];
    float sc2 = s2row[sidx];
    __builtin_prefetch(q1row + kb + 128, 0, 0);  // global_prefetch_b8, next slabs
    __builtin_prefetch(q2row + kb + 128, 0, 0);
    v16h b1 = dequant16(q1row + kb + koffB, tab, sc1);
    v16h b2 = dequant16(q2row + kb + koffB, tab, sc2);
    __syncthreads();

#pragma unroll
    for (int s = 0; s < 4; ++s) {
      const _Float16* ap = &xs[(s * 16 + frag) * 32];
      v8h alo = *(const v8h*)(ap + koffB);       // ds_load_b128
      v8h ahi = *(const v8h*)(ap + koffB + 8);
      v16h a;
#pragma unroll
      for (int j = 0; j < 8; ++j) { a[j] = alo[j]; a[8 + j] = ahi[j]; }
      acc1[s] = __builtin_amdgcn_wmma_f32_16x16x32_f16(
          false, a, false, b1, (short)0, acc1[s], false, false);
      acc2[s] = __builtin_amdgcn_wmma_f32_16x16x32_f16(
          false, a, false, b2, (short)0, acc2[s], false, false);
    }
  }

  // Epilogue: silu(gate) * up, f16 store. Fast reciprocal (v_rcp_f32) instead of
  // the IEEE div expansion -- activation precision is ample at 1 ulp rcp.
  // C layout: VGPR r -> row r (lanes 0-15) / row r+8 (lanes 16-31), col = lane&15.
  const int col = n0 + frag;
#pragma unroll
  for (int s = 0; s < 4; ++s) {
#pragma unroll
    for (int r = 0; r < 8; ++r) {
      float g = acc1[s][r];
      float u = acc2[s][r];
      float sig = __builtin_amdgcn_rcpf(1.0f + __expf(-g));
      float hv = g * sig * u;
      int row = m0 + s * 16 + kgrp * 8 + r;
      hout[(size_t)row * HIDDEN + col] = (_Float16)hv;
    }
  }
}

// ---------------------------------------------------------------------------
// Kernel 2: out[M,D] = h @ w3^T  (A = f16 h, B = dequantized w3, K = HIDDEN)
// The f16 h tile is staged with CDNA5 async global->LDS copies (no VGPR
// round-trip), synchronized with s_wait_asynccnt before the barrier.
// ---------------------------------------------------------------------------
__global__ __launch_bounds__(128) void nf4_down_kernel(
    const _Float16* __restrict__ hin,  // [M, H] f16
    const int* __restrict__ w3q,       // [D, H] codes
    const float* __restrict__ w3s,     // [D, H/64]
    float* __restrict__ out)           // [M, D] f32
{
  __shared__ _Float16 hs[64 * 32];
  __shared__ float    tab[16];

  const int tid  = threadIdx.x;
  const int lane = tid & 31;
  const int wave = tid >> 5;
  if (tid < 16) tab[tid] = NF4_TAB[tid];

  const int m0 = blockIdx.y * 64;
  const int n0 = blockIdx.x * 64 + wave * 16;

  const int frag  = lane & 15;
  const int kgrp  = lane >> 4;
  const int koffA = kgrp * 8;
  const int koffB = kgrp * 16;

  const int*   q3row = w3q + (size_t)(n0 + frag) * HIDDEN;
  const float* s3row = w3s + (size_t)(n0 + frag) * (HIDDEN / QBLK);

  v8f acc[4];
  v8f vz = {};
#pragma unroll
  for (int s = 0; s < 4; ++s) acc[s] = vz;

  const int sr = tid >> 1;
  const int sc = (tid & 1) * 16;
  const _Float16* hsrc = hin + (size_t)(m0 + sr) * HIDDEN + sc;
  // LDS byte address for this thread's 32-byte staging slot. Flat->LDS mapping
  // keeps the LDS offset in addr[31:0] (ISA 10.2), so truncation is the offset.
  const unsigned lds_dst =
      (unsigned)(uintptr_t)(&hs[sr * 32 + sc]);

  for (int kb = 0; kb < HIDDEN; kb += 32) {
    __syncthreads();
    {
      const _Float16* g0 = hsrc + kb;      // 16B chunk 0
      const _Float16* g1 = hsrc + kb + 8;  // 16B chunk 1
      // GLOBAL_LOAD_ASYNC_TO_LDS_B128, GV mode: vdst = LDS addr, v[a:a+1] = addr
      asm volatile("global_load_async_to_lds_b128 %0, %1, off"
                   :: "v"(lds_dst), "v"(g0) : "memory");
      asm volatile("global_load_async_to_lds_b128 %0, %1, off"
                   :: "v"(lds_dst + 16u), "v"(g1) : "memory");
    }

    // Dequantize the w3 B-fragment while the async DMA fills LDS.
    const int sidx = (kb + koffB) >> 6;
    float sc3 = s3row[sidx];
    __builtin_prefetch(q3row + kb + 128, 0, 0);
    v16h b = dequant16(q3row + kb + koffB, tab, sc3);

    asm volatile("s_wait_asynccnt 0x0" ::: "memory");
    __syncthreads();

#pragma unroll
    for (int s = 0; s < 4; ++s) {
      const _Float16* ap = &hs[(s * 16 + frag) * 32];
      v8h alo = *(const v8h*)(ap + koffB);
      v8h ahi = *(const v8h*)(ap + koffB + 8);
      v16h a;
#pragma unroll
      for (int j = 0; j < 8; ++j) { a[j] = alo[j]; a[8 + j] = ahi[j]; }
      acc[s] = __builtin_amdgcn_wmma_f32_16x16x32_f16(
          false, a, false, b, (short)0, acc[s], false, false);
    }
  }

  const int col = n0 + frag;
#pragma unroll
  for (int s = 0; s < 4; ++s)
#pragma unroll
    for (int r = 0; r < 8; ++r) {
      int row = m0 + s * 16 + kgrp * 8 + r;
      out[(size_t)row * D_MODEL + col] = acc[s][r];
    }
}

// ---------------------------------------------------------------------------
// Harness entry point.
// d_in order (setup_inputs): x, w1_q, w1_scale, w2_q, w2_scale, w3_q, w3_scale
// d_ws: holds the f16 intermediate h [MTOT, HIDDEN] = 90.2 MB.
// ---------------------------------------------------------------------------
extern "C" void kernel_launch(void* const* d_in, const int* in_sizes, int n_in,
                              void* d_out, int out_size, void* d_ws, size_t ws_size,
                              hipStream_t stream) {
  (void)in_sizes; (void)n_in; (void)out_size; (void)ws_size;
  const float* x   = (const float*)d_in[0];
  const int*   w1q = (const int*)d_in[1];
  const float* w1s = (const float*)d_in[2];
  const int*   w2q = (const int*)d_in[3];
  const float* w2s = (const float*)d_in[4];
  const int*   w3q = (const int*)d_in[5];
  const float* w3s = (const float*)d_in[6];
  float*       out = (float*)d_out;
  _Float16*    hbuf = (_Float16*)d_ws;   // MTOT*HIDDEN f16

  dim3 blk(128);
  dim3 g1(HIDDEN / 64, MTOT / 64);       // 172 x 64
  nf4_gateup_kernel<<<g1, blk, 0, stream>>>(x, w1q, w1s, w2q, w2s, hbuf);

  dim3 g2(D_MODEL / 64, MTOT / 64);      // 64 x 64
  nf4_down_kernel<<<g2, blk, 0, stream>>>(hbuf, w3q, w3s, out);
}
